// E2E_27479200760302
// MI455X (gfx1250) — compile-verified, matrix operand order checked by reading, not run on hardware
//
#include <hip/hip_runtime.h>
#include <cstddef>
#include <cstdint>

// ---------------- types ----------------
typedef __bf16 bf16_t;
typedef __attribute__((ext_vector_type(16))) __bf16 v16bf;
typedef __attribute__((ext_vector_type(8)))  float  v8f;

#define B_   32
#define T_   512
#define E_   1024
#define H_   1024
#define A_   512
#define V_   16000
#define L_   63
#define OLEN_ 64
#define HE_  2048       // H+E
#define G4H_ 4096       // 4*H
#define SOS_ 1
#define EOS_ 2

// ---------------- WMMA helpers (CDNA5 16x16x32 bf16, f32 accum) ----------------
__device__ __forceinline__ v8f wmma_bf16(v16bf a, v16bf b, v8f c) {
  // (neg_a, A, neg_b, B, c_mod, C, reuse_a, reuse_b)
  return __builtin_amdgcn_wmma_f32_16x16x32_bf16(false, a, false, b, (short)0, c, false, false);
}

union frag16 {
  v16bf v;
  uint4 q[2];
};

// 16x32 (RxK) fragment from a row-major [*, ld] bf16 matrix.
// ISA layout: lane holds row r = lane&15, half = lane>>4.
// VGPR v (v<4):  k = 2v + 8*half        -> dwords row32[4*half + v]
// VGPR v (v>=4): k = 16 + 2(v-4)+8*half -> dwords row32[8 + 4*half + (v-4)]
// => two contiguous 16B runs: uint4 at index half and 2+half.
__device__ __forceinline__ v16bf load_frag(const bf16_t* __restrict__ M, int ld,
                                           int rbase, int kbase) {
  int lane = threadIdx.x & 31;
  int r = rbase + (lane & 15);
  int half = lane >> 4;
  const uint4* p = (const uint4*)(M + (size_t)r * ld + kbase);
  frag16 f;
  f.q[0] = p[half];
  f.q[1] = p[2 + half];
  return f.v;
}

__device__ __forceinline__ float sigmoidf_(float x) { return 1.0f / (1.0f + __expf(-x)); }

// ---------------- setup kernels ----------------
__global__ void f2bf_kernel(const float* __restrict__ in, bf16_t* __restrict__ out, size_t n) {
  size_t i = (size_t)blockIdx.x * 256 + threadIdx.x;
  if (i < n) out[i] = (bf16_t)in[i];
}

// transpose + convert: in [R, C] f32 -> out [C, R] bf16 (LDS-tiled)
__global__ void f2bf_transpose_kernel(const float* __restrict__ in,
                                      bf16_t* __restrict__ out, int R, int C) {
  __shared__ float tile[32][33];
  int c0 = blockIdx.x * 32, r0 = blockIdx.y * 32;
  int tx = threadIdx.x & 31, ty = threadIdx.x >> 5;   // 32 x 8
  for (int i = ty; i < 32; i += 8) {
    int r = r0 + i, c = c0 + tx;
    tile[i][tx] = (r < R && c < C) ? in[(size_t)r * C + c] : 0.f;
  }
  __syncthreads();
  for (int i = ty; i < 32; i += 8) {
    int c = c0 + i, r = r0 + tx;
    if (c < C && r < R) out[(size_t)c * R + r] = (bf16_t)tile[tx][i];
  }
}

__global__ void mask_hpad_kernel(const float* __restrict__ hpad, const int* __restrict__ hlen,
                                 bf16_t* __restrict__ out) {
  size_t i = (size_t)blockIdx.x * 256 + threadIdx.x;
  if (i >= (size_t)B_ * T_ * E_) return;
  int b = (int)(i >> 19);
  int t = (int)((i >> 10) & (T_ - 1));
  float v = (t < hlen[b]) ? hpad[i] : 0.0f;
  out[i] = (bf16_t)v;
}

__global__ void embed_kernel(const int* __restrict__ ys, const float* __restrict__ embed_W,
                             bf16_t* __restrict__ eys) {
  int row = blockIdx.x;                 // b*OLEN + t
  int b = row >> 6, t = row & 63;
  int tok = (t == 0) ? SOS_ : ys[b * L_ + (t - 1)];
  const float* src = embed_W + (size_t)tok * H_;
  bf16_t* dst = eys + (size_t)row * H_;
  for (int i = threadIdx.x; i < H_; i += 256) dst[i] = (bf16_t)src[i];
}

__global__ void init_state_kernel(float* z0, float* c0, float* z1, float* c1,
                                  bf16_t* z0b, bf16_t* z1b) {
  int i = blockIdx.x * 256 + threadIdx.x;
  if (i < B_ * H_) {
    z0[i] = 0.f; c0[i] = 0.f; z1[i] = 0.f; c1[i] = 0.f;
    z0b[i] = (bf16_t)0.f; z1b[i] = (bf16_t)0.f;
  }
}

// ---------------- pre_enc = tanh(hpad_bf @ W_enc + b_enc) ----------------
// A: hpad_bf [16384,1024]; B: WencT [512,1024] (n,k). One wave per 16x16 tile.
__global__ void preenc_kernel(const bf16_t* __restrict__ hpad_bf,
                              const bf16_t* __restrict__ WencT,
                              const float* __restrict__ b_enc,
                              float* __restrict__ preenc) {        // [16384,512]
  int wid = blockIdx.x * 8 + (threadIdx.x >> 5);
  int mtile = wid >> 5;        // 1024 m-tiles
  int ntile = wid & 31;        // 32 n-tiles
  v8f c = {};
  for (int k = 0; k < E_; k += 32) {
    v16bf a = load_frag(hpad_bf, E_, mtile * 16, k);
    v16bf b = load_frag(WencT, E_, ntile * 16, k);
    c = wmma_bf16(a, b, c);
  }
  int lane = threadIdx.x & 31;
  int n = ntile * 16 + (lane & 15);
  int half = lane >> 4;
  float bias = b_enc[n];
#pragma unroll
  for (int v = 0; v < 8; ++v) {
    int m = mtile * 16 + v + 8 * half;
    preenc[(size_t)m * A_ + n] = tanhf(c[v] + bias);
  }
}

// ---------------- per-step attention (VALU fp32, one block per batch row) ----------------
__global__ void att_step_kernel(const float* __restrict__ preenc,  // [B*T, A]
                                const float* __restrict__ hpad,    // [B,T,E]
                                const int* __restrict__ hlen,
                                const float* __restrict__ W_dec,   // [H, A]
                                const float* __restrict__ b_dec,   // [A]
                                const float* __restrict__ z0,      // [B, H]
                                const bf16_t* __restrict__ eys,    // [B,OLEN,H]
                                int tstep,
                                float* __restrict__ att_c,         // [B, E]
                                bf16_t* __restrict__ x_bf,         // [B, HE]
                                bf16_t* __restrict__ z_all) {      // [B,OLEN,HE]
  int b = blockIdx.x;
  int tid = threadIdx.x;
  __shared__ float s_z0[H_];
  __shared__ float s_dec[A_];
  __shared__ float s_w[T_];
  __shared__ float s_red[256];

  for (int i = tid; i < H_; i += 256) s_z0[i] = z0[b * H_ + i];
  __syncthreads();

  for (int a = tid; a < A_; a += 256) {
    float acc = b_dec[a];
    for (int k = 0; k < H_; ++k) acc += s_z0[k] * W_dec[(size_t)k * A_ + a];
    s_dec[a] = tanhf(acc);
  }
  __syncthreads();

  int len = hlen[b];
  for (int t = tid; t < T_; t += 256) {
    const float* pe = preenc + ((size_t)b * T_ + t) * A_;
    float acc = 0.f;
    for (int a = 0; a < A_; ++a) acc += pe[a] * s_dec[a];
    s_w[t] = (t < len) ? acc : -1e9f;
  }
  __syncthreads();

  float m = -INFINITY;
  for (int t = tid; t < T_; t += 256) m = fmaxf(m, s_w[t]);
  s_red[tid] = m; __syncthreads();
  for (int s = 128; s > 0; s >>= 1) { if (tid < s) s_red[tid] = fmaxf(s_red[tid], s_red[tid + s]); __syncthreads(); }
  float gmax = s_red[0]; __syncthreads();
  float p = 0.f;
  for (int t = tid; t < T_; t += 256) { float v = __expf(s_w[t] - gmax); s_w[t] = v; p += v; }
  s_red[tid] = p; __syncthreads();
  for (int s = 128; s > 0; s >>= 1) { if (tid < s) s_red[tid] += s_red[tid + s]; __syncthreads(); }
  float inv = 1.0f / s_red[0];
  __syncthreads();

  const float* hp = hpad + (size_t)b * T_ * E_;
  float a0 = 0.f, a1 = 0.f, a2 = 0.f, a3 = 0.f;
  for (int t = 0; t < T_; ++t) {
    float w = s_w[t];
    const float* row = hp + (size_t)t * E_;
    a0 += w * row[tid];
    a1 += w * row[tid + 256];
    a2 += w * row[tid + 512];
    a3 += w * row[tid + 768];
  }
  float outv[4] = {a0 * inv, a1 * inv, a2 * inv, a3 * inv};
  int   offs[4] = {tid, tid + 256, tid + 512, tid + 768};
  size_t zbase = ((size_t)b * OLEN_ + tstep) * HE_;
#pragma unroll
  for (int j = 0; j < 4; ++j) {
    int e = offs[j];
    att_c[b * E_ + e] = outv[j];
    bf16_t h = (bf16_t)outv[j];
    x_bf[b * HE_ + H_ + e] = h;
    z_all[zbase + H_ + e] = h;
  }
  const bf16_t* ey = eys + ((size_t)b * OLEN_ + tstep) * H_;
  for (int i = tid; i < H_; i += 256) x_bf[b * HE_ + i] = ey[i];
}

// ---------------- LSTM gates GEMM: gates = X @ Wih^T + Hst @ Whh^T ----------------
__global__ void lstm_gates_kernel(const bf16_t* __restrict__ X,    // [32, Ka]
                                  const bf16_t* __restrict__ Wih,  // [4096, Ka]
                                  int Ka,
                                  const bf16_t* __restrict__ Hst,  // [32, Kh]
                                  const bf16_t* __restrict__ Whh,  // [4096, Kh]
                                  int Kh,
                                  float* __restrict__ gates) {     // [32, 4096]
  int wid = blockIdx.x * 8 + (threadIdx.x >> 5);
  int mtile = wid >> 8;        // 0..1
  int ntile = wid & 255;       // 0..255
  int mbase = mtile * 16, nbase = ntile * 16;
  v8f c = {};
  for (int k = 0; k < Ka; k += 32) {
    v16bf a = load_frag(X, Ka, mbase, k);
    v16bf b = load_frag(Wih, Ka, nbase, k);
    c = wmma_bf16(a, b, c);
  }
  for (int k = 0; k < Kh; k += 32) {
    v16bf a = load_frag(Hst, Kh, mbase, k);
    v16bf b = load_frag(Whh, Kh, nbase, k);
    c = wmma_bf16(a, b, c);
  }
  int lane = threadIdx.x & 31;
  int n = nbase + (lane & 15);
  int half = lane >> 4;
#pragma unroll
  for (int v = 0; v < 8; ++v) {
    int m = mbase + v + 8 * half;
    gates[(size_t)m * G4H_ + n] = c[v];
  }
}

// ---------------- LSTM pointwise (i,f,g,o) ----------------
__global__ void lstm_elem_kernel(const float* __restrict__ gates,
                                 const float* __restrict__ b_ih,
                                 const float* __restrict__ b_hh,
                                 float* __restrict__ c_state,
                                 float* __restrict__ h_state,
                                 bf16_t* __restrict__ h_bf,
                                 bf16_t* __restrict__ z_all,
                                 int tstep) {
  int idx = blockIdx.x * 256 + threadIdx.x;
  if (idx >= B_ * H_) return;
  int b = idx >> 10, j = idx & (H_ - 1);
  const float* g = gates + (size_t)b * G4H_;
  float gi = g[j]          + b_ih[j]          + b_hh[j];
  float gf = g[H_ + j]     + b_ih[H_ + j]     + b_hh[H_ + j];
  float gg = g[2 * H_ + j] + b_ih[2 * H_ + j] + b_hh[2 * H_ + j];
  float go = g[3 * H_ + j] + b_ih[3 * H_ + j] + b_hh[3 * H_ + j];
  float cn = sigmoidf_(gf) * c_state[idx] + sigmoidf_(gi) * tanhf(gg);
  float hn = sigmoidf_(go) * tanhf(cn);
  c_state[idx] = cn;
  h_state[idx] = hn;
  bf16_t hb = (bf16_t)hn;
  h_bf[idx] = hb;
  if (z_all) z_all[((size_t)b * OLEN_ + tstep) * HE_ + j] = hb;
}

// ---------------- fused output projection + online log-softmax NLL ----------------
// One block per 16-row tile. 8 waves split 500 column-tile PAIRS (2-wide N blocking:
// one A fragment feeds two WMMAs). Online (max,sumexp) per row in registers.
__global__ void out_ce_kernel(const bf16_t* __restrict__ Z,      // [2048, 2048]
                              const bf16_t* __restrict__ WoutT,  // [16000, 2048] (n,k)
                              const float* __restrict__ b_out,
                              const int* __restrict__ ys,        // [32, 63]
                              float* __restrict__ nll) {         // [2048]
  __shared__ float red_max[8 * 32 * 8];
  __shared__ float red_sum[8 * 32 * 8];
  __shared__ float s_lab[16];
  __shared__ int   s_label[16];
  int rb = blockIdx.x;
  int tid = threadIdx.x;
  int wave = tid >> 5, lane = tid & 31;
  int half = lane >> 4;

  if (tid < 16) {
    int r = rb * 16 + tid;
    int b = r >> 6, t = r & 63;
    s_label[tid] = (t < L_) ? ys[b * L_ + t] : EOS_;
    s_lab[tid] = 0.f;
  }
  __syncthreads();

  int labv[8];
#pragma unroll
  for (int v = 0; v < 8; ++v) labv[v] = s_label[v + 8 * half];

  int mbase = rb * 16;
  float pmax[8], psum[8];
#pragma unroll
  for (int v = 0; v < 8; ++v) { pmax[v] = -INFINITY; psum[v] = 0.f; }

  // 500 pairs of column tiles over 8 waves
  for (int pr = wave; pr < (V_ / 32); pr += 8) {
    int nb0 = pr * 32;        // first tile base
    int nb1 = nb0 + 16;       // second tile base
    if (pr + 8 < (V_ / 32))   // prefetch next pair's rows (global_prefetch_b8)
      __builtin_prefetch(WoutT + ((size_t)(pr + 8) * 32 + (lane & 31)) * HE_, 0, 1);
    v8f c0 = {}, c1 = {};
    for (int k = 0; k < HE_; k += 32) {
      v16bf a  = load_frag(Z, HE_, mbase, k);
      v16bf b0 = load_frag(WoutT, HE_, nb0, k);
      c0 = wmma_bf16(a, b0, c0);
      v16bf b1 = load_frag(WoutT, HE_, nb1, k);
      c1 = wmma_bf16(a, b1, c1);
    }
    int n0 = nb0 + (lane & 15);
    int n1 = nb1 + (lane & 15);
    float bias0 = b_out[n0];
    float bias1 = b_out[n1];
#pragma unroll
    for (int v = 0; v < 8; ++v) {
      float val0 = c0[v] + bias0;
      float val1 = c1[v] + bias1;
      if (n0 == labv[v]) s_lab[v + 8 * half] = val0;
      if (n1 == labv[v]) s_lab[v + 8 * half] = val1;
      float nm = fmaxf(pmax[v], fmaxf(val0, val1));
      psum[v] = psum[v] * __expf(pmax[v] - nm) + __expf(val0 - nm) + __expf(val1 - nm);
      pmax[v] = nm;
    }
  }
#pragma unroll
  for (int v = 0; v < 8; ++v) {
    red_max[(wave * 32 + lane) * 8 + v] = pmax[v];
    red_sum[(wave * 32 + lane) * 8 + v] = psum[v];
  }
  __syncthreads();

  if (tid < 16) {
    int m = tid;
    int vslot = m & 7, hf = m >> 3;
    float M = -INFINITY, S = 0.f;
    for (int w = 0; w < 8; ++w)
      for (int l = 0; l < 16; ++l) {
        int lane2 = hf * 16 + l;
        float pm = red_max[(w * 32 + lane2) * 8 + vslot];
        float ps = red_sum[(w * 32 + lane2) * 8 + vslot];
        float nm = fmaxf(M, pm);
        S = S * __expf(M - nm) + ps * __expf(pm - nm);
        M = nm;
      }
    nll[rb * 16 + m] = (M + __logf(S)) - s_lab[m];
  }
}

__global__ void loss_kernel(const float* __restrict__ nll, float* __restrict__ out) {
  __shared__ float red[256];
  int tid = threadIdx.x;
  float s = 0.f;
  for (int i = tid; i < B_ * OLEN_; i += 256) s += nll[i];
  red[tid] = s; __syncthreads();
  for (int st = 128; st > 0; st >>= 1) { if (tid < st) red[tid] += red[tid + st]; __syncthreads(); }
  if (tid == 0) out[0] = red[0] / (float)(B_ * OLEN_) * (float)(OLEN_ - 1);
}

// ---------------- host ----------------
extern "C" void kernel_launch(void* const* d_in, const int* in_sizes, int n_in,
                              void* d_out, int out_size, void* d_ws, size_t ws_size,
                              hipStream_t stream) {
  (void)in_sizes; (void)n_in; (void)out_size; (void)ws_size;
  const float* hpad    = (const float*)d_in[0];
  const int*   hlen    = (const int*)d_in[1];
  const int*   ys      = (const int*)d_in[2];
  const float* embed_W = (const float*)d_in[3];
  const float* W_enc   = (const float*)d_in[4];
  const float* b_enc   = (const float*)d_in[5];
  const float* W_dec   = (const float*)d_in[6];
  const float* b_dec   = (const float*)d_in[7];
  const float* W_ih0   = (const float*)d_in[8];
  const float* W_hh0   = (const float*)d_in[9];
  const float* b_ih0   = (const float*)d_in[10];
  const float* b_hh0   = (const float*)d_in[11];
  const float* W_ih1   = (const float*)d_in[12];
  const float* W_hh1   = (const float*)d_in[13];
  const float* b_ih1   = (const float*)d_in[14];
  const float* b_hh1   = (const float*)d_in[15];
  const float* W_out   = (const float*)d_in[16];
  const float* b_out   = (const float*)d_in[17];
  float* out = (float*)d_out;

  char* wp = (char*)d_ws;
  auto alloc = [&](size_t bytes) -> char* {
    char* p = wp;
    wp += (bytes + 255) & ~(size_t)255;
    return p;
  };
  bf16_t* hpad_bf = (bf16_t*)alloc((size_t)B_ * T_ * E_ * 2);
  bf16_t* WencT   = (bf16_t*)alloc((size_t)A_ * E_ * 2);       // [512,1024]
  bf16_t* Wih0_bf = (bf16_t*)alloc((size_t)G4H_ * HE_ * 2);
  bf16_t* Whh0_bf = (bf16_t*)alloc((size_t)G4H_ * H_ * 2);
  bf16_t* Wih1_bf = (bf16_t*)alloc((size_t)G4H_ * H_ * 2);
  bf16_t* Whh1_bf = (bf16_t*)alloc((size_t)G4H_ * H_ * 2);
  bf16_t* WoutT   = (bf16_t*)alloc((size_t)V_ * HE_ * 2);      // [16000,2048]
  bf16_t* eys_bf  = (bf16_t*)alloc((size_t)B_ * OLEN_ * H_ * 2);
  bf16_t* z_all   = (bf16_t*)alloc((size_t)B_ * OLEN_ * HE_ * 2);
  float*  preenc  = (float*)alloc((size_t)B_ * T_ * A_ * 4);
  float*  z0      = (float*)alloc((size_t)B_ * H_ * 4);
  float*  c0      = (float*)alloc((size_t)B_ * H_ * 4);
  float*  z1      = (float*)alloc((size_t)B_ * H_ * 4);
  float*  c1      = (float*)alloc((size_t)B_ * H_ * 4);
  bf16_t* z0_bf   = (bf16_t*)alloc((size_t)B_ * H_ * 2);
  bf16_t* z1_bf   = (bf16_t*)alloc((size_t)B_ * H_ * 2);
  bf16_t* x_bf    = (bf16_t*)alloc((size_t)B_ * HE_ * 2);
  float*  att_c   = (float*)alloc((size_t)B_ * E_ * 4);
  float*  gates0  = (float*)alloc((size_t)B_ * G4H_ * 4);
  float*  gates1  = (float*)alloc((size_t)B_ * G4H_ * 4);
  float*  nll     = (float*)alloc((size_t)B_ * OLEN_ * 4);

  auto cvt = [&](const float* src, bf16_t* dst, size_t n) {
    f2bf_kernel<<<(unsigned)((n + 255) / 256), 256, 0, stream>>>(src, dst, n);
  };
  // LSTM weights: already [N,K] for x @ W^T — plain convert
  cvt(W_ih0, Wih0_bf, (size_t)G4H_ * HE_);
  cvt(W_hh0, Whh0_bf, (size_t)G4H_ * H_);
  cvt(W_ih1, Wih1_bf, (size_t)G4H_ * H_);
  cvt(W_hh1, Whh1_bf, (size_t)G4H_ * H_);
  // W_enc [E,A] -> WencT [A,E] ; W_out [HE,V] -> WoutT [V,HE]
  {
    dim3 g(A_ / 32, E_ / 32);
    f2bf_transpose_kernel<<<g, 256, 0, stream>>>(W_enc, WencT, E_, A_);
  }
  {
    dim3 g(V_ / 32, HE_ / 32);
    f2bf_transpose_kernel<<<g, 256, 0, stream>>>(W_out, WoutT, HE_, V_);
  }

  {
    size_t n = (size_t)B_ * T_ * E_;
    mask_hpad_kernel<<<(unsigned)((n + 255) / 256), 256, 0, stream>>>(hpad, hlen, hpad_bf);
  }
  embed_kernel<<<B_ * OLEN_, 256, 0, stream>>>(ys, embed_W, eys_bf);
  init_state_kernel<<<(B_ * H_ + 255) / 256, 256, 0, stream>>>(z0, c0, z1, c1, z0_bf, z1_bf);

  preenc_kernel<<<4096, 256, 0, stream>>>(hpad_bf, WencT, b_enc, preenc);

  for (int t = 0; t < OLEN_; ++t) {
    att_step_kernel<<<B_, 256, 0, stream>>>(preenc, hpad, hlen, W_dec, b_dec, z0,
                                            eys_bf, t, att_c, x_bf, z_all);
    lstm_gates_kernel<<<64, 256, 0, stream>>>(x_bf, Wih0_bf, HE_, z0_bf, Whh0_bf, H_, gates0);
    lstm_elem_kernel<<<(B_ * H_ + 255) / 256, 256, 0, stream>>>(
        gates0, b_ih0, b_hh0, c0, z0, z0_bf, (bf16_t*)nullptr, t);
    lstm_gates_kernel<<<64, 256, 0, stream>>>(z0_bf, Wih1_bf, H_, z1_bf, Whh1_bf, H_, gates1);
    lstm_elem_kernel<<<(B_ * H_ + 255) / 256, 256, 0, stream>>>(
        gates1, b_ih1, b_hh1, c1, z1, z1_bf, z_all, t);
  }

  out_ce_kernel<<<(B_ * OLEN_) / 16, 256, 0, stream>>>(z_all, WoutT, b_out, ys, nll);
  loss_kernel<<<1, 256, 0, stream>>>(nll, out);
}
